// TransformerSeq2Seq_47347719471565
// MI455X (gfx1250) — matrix operand order repaired and checked by reading, not compiled, not run on hardware
//
#include <hip/hip_runtime.h>
#include <hip/hip_bf16.h>

// ---------------------------------------------------------------------------
// Transformer seq2seq forward for MI455X (gfx1250, wave32, WMMA).
// All GEMM-shaped math runs through v_wmma_f32_16x16x32_bf16 (bf16 operands,
// f32 accumulate). Activations stay f32 for residual/LayerNorm numerics.
// ---------------------------------------------------------------------------

typedef __attribute__((ext_vector_type(16))) __bf16 v16bf;
typedef __attribute__((ext_vector_type(8)))  __bf16 v8bf;
typedef __attribute__((ext_vector_type(8)))  float  v8f;

constexpr int V_   = 32000;
constexpr int D_   = 512;
constexpr int H_   = 8;
constexpr int DFF_ = 2048;
constexpr int NE_  = 6;
constexpr int ND_  = 6;
constexpr int S_   = 512;   // src len (== tgt len T)
constexpr int B_   = 8;
constexpr int DK_  = 64;    // D/H
constexpr int NTOK = S_ * B_;   // 4096 token rows
constexpr int BH_  = B_ * H_;   // 64

// ---------------------------------------------------------------------------
// Embedding gather + sqrt(D) scale + sinusoidal positional encoding.
// ---------------------------------------------------------------------------
__global__ __launch_bounds__(256) void embed_kernel(
    const int* __restrict__ toks, const float* __restrict__ emb,
    float* __restrict__ out)
{
  int gid   = blockIdx.x * 256 + threadIdx.x;
  int token = gid >> 9;          // / D_
  int dcol  = gid & (D_ - 1);
  int s     = token / B_;        // sequence position
  int t     = toks[token];
  int i2    = (dcol >> 1) * 2;
  float freq = __expf(-9.210340371976184f * (float)i2 * (1.0f / (float)D_)); // ln(10000)
  float arg  = (float)s * freq;
  float pe   = (dcol & 1) ? __cosf(arg) : __sinf(arg);
  out[gid] = emb[(size_t)t * D_ + dcol] * 22.627416997969522f + pe; // sqrt(512)
}

// ---------------------------------------------------------------------------
// Generic WMMA GEMM: C[M,N] = act(A[M,K] @ W[K,N] + bias)
// Block = 256 threads = 8 waves; block tile = 128(M) x 64(N); K step = 32.
// Wave w: wm = w&3 -> 32-row band, wn = w>>2 -> 32-col band; 4 WMMAs/step.
// LDS: A row-major (fragment = two contiguous 16B chunks per lane),
//      B transposed [n][k] (fragment = two 16B chunks per lane).
// Double-buffered, software-pipelined: gload(kt+1) | wmma(kt) | lstore(kt+1).
// ---------------------------------------------------------------------------
__global__ __launch_bounds__(256) void gemm_bias_kernel(
    const float* __restrict__ A, const float* __restrict__ W,
    const float* __restrict__ bias, float* __restrict__ C,
    int M, int N, int K, int relu)
{
  __shared__ __bf16 As[2][128 * 32];
  __shared__ __bf16 BsT[2][64 * 32];   // [n][k]

  const int tid  = threadIdx.x;
  const int lane = tid & 31;
  const int wave = tid >> 5;
  const int wm   = wave & 3;
  const int wn   = wave >> 2;
  const int hh   = lane >> 4;
  const int l16  = lane & 15;
  const int m0   = blockIdx.y * 128;
  const int n0   = blockIdx.x * 64;

  // cooperative staging indices
  const int ar = tid >> 1;            // 0..127
  const int ac = (tid & 1) * 16;      // 0,16
  const int br = tid >> 3;            // 0..31  (k row)
  const int bc = (tid & 7) * 8;       // 0..56  (n col)

  const int KT = K >> 5;

  float4 ra[4], rb[2];

  auto gload = [&](int kt) {
    const float4* pa = (const float4*)(A + (size_t)(m0 + ar) * K + kt * 32 + ac);
    ra[0] = pa[0]; ra[1] = pa[1]; ra[2] = pa[2]; ra[3] = pa[3];
    const float4* pb = (const float4*)(W + (size_t)(kt * 32 + br) * N + n0 + bc);
    rb[0] = pb[0]; rb[1] = pb[1];
    if (kt + 2 < KT) {  // keep L2 path warm for the tile after next
      __builtin_prefetch(A + (size_t)(m0 + ar) * K + (kt + 2) * 32 + ac, 0, 1);
      __builtin_prefetch(W + (size_t)((kt + 2) * 32 + br) * N + n0 + bc, 0, 1);
    }
  };

  auto lstore = [&](int buf) {
    __bf16* dstA = &As[buf][ar * 32 + ac];
    const float* fa = (const float*)ra;
    #pragma unroll
    for (int j = 0; j < 16; ++j) dstA[j] = (__bf16)fa[j];   // contiguous -> wide ds stores
    const float* fb = (const float*)rb;
    #pragma unroll
    for (int j = 0; j < 8; ++j) BsT[buf][(bc + j) * 32 + br] = (__bf16)fb[j];
  };

  // A fragment: lane row = l16; elements a[0..7]=k(8*hh..), a[8..15]=k(16+8*hh..)
  auto load_a = [&](const __bf16* rowbase) -> v16bf {
    const v8bf* p = (const v8bf*)rowbase;
    v8bf lo = p[hh];
    v8bf hi = p[2 + hh];
    v16bf r;
    #pragma unroll
    for (int e = 0; e < 8; ++e) { r[e] = lo[e]; r[8 + e] = hi[e]; }
    return r;
  };
  // B fragment: lane col = l16; elements b[0..7]=k(8*hh..), b[8..15]=k(16+8*hh..)
  auto load_b = [&](const __bf16* colbase) -> v16bf {
    const v8bf* p = (const v8bf*)colbase;
    v8bf lo = p[hh];
    v8bf hi = p[2 + hh];
    v16bf r;
    #pragma unroll
    for (int e = 0; e < 8; ++e) { r[e] = lo[e]; r[8 + e] = hi[e]; }
    return r;
  };

  const v8f z = {0.f, 0.f, 0.f, 0.f, 0.f, 0.f, 0.f, 0.f};
  v8f acc00 = z, acc01 = z, acc10 = z, acc11 = z;

  gload(0);
  lstore(0);
  __syncthreads();

  for (int kt = 0; kt < KT; ++kt) {
    const int  buf  = kt & 1;
    const bool more = (kt + 1 < KT);
    if (more) gload(kt + 1);          // global loads overlap WMMAs below

    const __bf16* Ab = As[buf];
    const __bf16* Bb = BsT[buf];
    v16bf a0 = load_a(Ab + (wm * 32 + l16) * 32);
    v16bf a1 = load_a(Ab + (wm * 32 + 16 + l16) * 32);
    v16bf b0 = load_b(Bb + (wn * 32 + l16) * 32);
    v16bf b1 = load_b(Bb + (wn * 32 + 16 + l16) * 32);

    acc00 = __builtin_amdgcn_wmma_f32_16x16x32_bf16(false, a0, false, b0, (short)0, acc00, false, false);
    acc01 = __builtin_amdgcn_wmma_f32_16x16x32_bf16(false, a0, false, b1, (short)0, acc01, false, false);
    acc10 = __builtin_amdgcn_wmma_f32_16x16x32_bf16(false, a1, false, b0, (short)0, acc10, false, false);
    acc11 = __builtin_amdgcn_wmma_f32_16x16x32_bf16(false, a1, false, b1, (short)0, acc11, false, false);

    if (more) lstore(buf ^ 1);        // fill the other buffer
    __syncthreads();                  // single barrier per K-step
  }

  const int n_lo = n0 + wn * 32 + l16;
  const int n_hi = n_lo + 16;
  const float bv0 = bias ? bias[n_lo] : 0.f;
  const float bv1 = bias ? bias[n_hi] : 0.f;
  #pragma unroll
  for (int r = 0; r < 8; ++r) {
    const int m_lo = m0 + wm * 32 + r + 8 * hh;
    const int m_hi = m_lo + 16;
    float v00 = acc00[r] + bv0, v01 = acc01[r] + bv1;
    float v10 = acc10[r] + bv0, v11 = acc11[r] + bv1;
    if (relu) {
      v00 = fmaxf(v00, 0.f); v01 = fmaxf(v01, 0.f);
      v10 = fmaxf(v10, 0.f); v11 = fmaxf(v11, 0.f);
    }
    C[(size_t)m_lo * N + n_lo] = v00;
    C[(size_t)m_lo * N + n_hi] = v01;
    C[(size_t)m_hi * N + n_lo] = v10;
    C[(size_t)m_hi * N + n_hi] = v11;
  }
}

// ---------------------------------------------------------------------------
// scores[bh,q,k] = (Q[q,b,h,:] . K[k,b,h,:]) / sqrt(dk)  (+ causal mask)
// One wave per 16x16 score tile. Q,K stored [S,B,D] with head h at cols h*64.
// ---------------------------------------------------------------------------
__global__ __launch_bounds__(32) void attn_scores_kernel(
    const float* __restrict__ q, const float* __restrict__ kmat,
    float* __restrict__ scores, int causal)
{
  const int lane = threadIdx.x;
  const int l16  = lane & 15;
  const int hh   = lane >> 4;
  const int kt   = blockIdx.x;   // Sk/16
  const int qt   = blockIdx.y;   // Sq/16
  const int bh   = blockIdx.z;   // B*H
  const int b    = bh / H_;
  const int h    = bh % H_;

  v8f acc = {0.f, 0.f, 0.f, 0.f, 0.f, 0.f, 0.f, 0.f};
  #pragma unroll
  for (int kd0 = 0; kd0 < DK_; kd0 += 32) {
    v16bf a, bf;
    const float* arow = q + ((size_t)(qt * 16 + l16) * B_ + b) * D_ + h * DK_ + kd0;
    #pragma unroll
    for (int i = 0; i < 8; ++i) {
      int kk = 2 * i + 16 * hh;
      a[2 * i]     = (__bf16)arow[kk];
      a[2 * i + 1] = (__bf16)arow[kk + 1];
    }
    const float* brow = kmat + ((size_t)(kt * 16 + l16) * B_ + b) * D_ + h * DK_ + kd0 + 16 * hh;
    #pragma unroll
    for (int e = 0; e < 16; ++e) bf[e] = (__bf16)brow[e];
    acc = __builtin_amdgcn_wmma_f32_16x16x32_bf16(
        false, a, false, bf, (short)0, acc, false, false);
  }
  const float scale = 0.125f;  // 1/sqrt(64)
  #pragma unroll
  for (int r = 0; r < 8; ++r) {
    int m = qt * 16 + r + 8 * hh;
    int n = kt * 16 + l16;
    float vz = acc[r] * scale;
    if (causal && n > m) vz = -1e9f;
    scores[((size_t)bh * S_ + m) * S_ + n] = vz;
  }
}

// ---------------------------------------------------------------------------
// Row softmax over length S_ (512): one wave32 per row, 16 elems/lane.
// ---------------------------------------------------------------------------
__global__ __launch_bounds__(256) void softmax_kernel(float* __restrict__ scores)
{
  const int lane = threadIdx.x & 31;
  const int row  = blockIdx.x * 8 + (threadIdx.x >> 5);
  float* p = scores + (size_t)row * S_;
  float v[16];
  float mx = -3.4e38f;
  #pragma unroll
  for (int j = 0; j < 16; ++j) { v[j] = p[lane + 32 * j]; mx = fmaxf(mx, v[j]); }
  #pragma unroll
  for (int o = 16; o >= 1; o >>= 1) mx = fmaxf(mx, __shfl_xor(mx, o, 32));
  float sum = 0.f;
  #pragma unroll
  for (int j = 0; j < 16; ++j) { v[j] = __expf(v[j] - mx); sum += v[j]; }
  #pragma unroll
  for (int o = 16; o >= 1; o >>= 1) sum += __shfl_xor(sum, o, 32);
  float inv = 1.f / sum;
  #pragma unroll
  for (int j = 0; j < 16; ++j) p[lane + 32 * j] = v[j] * inv;
}

// ---------------------------------------------------------------------------
// ctx[q,b,h,:] = sum_k attn[bh,q,k] * V[k,b,h,:]   (one wave per 16x16 tile)
// ---------------------------------------------------------------------------
__global__ __launch_bounds__(32) void attn_ctx_kernel(
    const float* __restrict__ attn, const float* __restrict__ vmat,
    float* __restrict__ ctx)
{
  const int lane = threadIdx.x;
  const int l16  = lane & 15;
  const int hh   = lane >> 4;
  const int dt   = blockIdx.x;   // DK/16 = 4
  const int qt   = blockIdx.y;   // S/16
  const int bh   = blockIdx.z;
  const int b    = bh / H_;
  const int h    = bh % H_;

  v8f acc = {0.f, 0.f, 0.f, 0.f, 0.f, 0.f, 0.f, 0.f};
  for (int k0 = 0; k0 < S_; k0 += 32) {
    v16bf a, bf;
    const float* arow = attn + ((size_t)bh * S_ + qt * 16 + l16) * S_ + k0;
    #pragma unroll
    for (int i = 0; i < 8; ++i) {
      int kk = 2 * i + 16 * hh;
      a[2 * i]     = (__bf16)arow[kk];
      a[2 * i + 1] = (__bf16)arow[kk + 1];
    }
    #pragma unroll
    for (int e = 0; e < 16; ++e) {
      int krow = k0 + 16 * hh + e;
      bf[e] = (__bf16)vmat[((size_t)krow * B_ + b) * D_ + h * DK_ + dt * 16 + l16];
    }
    acc = __builtin_amdgcn_wmma_f32_16x16x32_bf16(
        false, a, false, bf, (short)0, acc, false, false);
  }
  #pragma unroll
  for (int r = 0; r < 8; ++r) {
    int m = qt * 16 + r + 8 * hh;
    ctx[((size_t)m * B_ + b) * D_ + h * DK_ + dt * 16 + l16] = acc[r];
  }
}

// ---------------------------------------------------------------------------
// x = LayerNorm(x + delta) * g + beta.  One wave32 per token row (D=512).
// ---------------------------------------------------------------------------
__global__ __launch_bounds__(256) void add_ln_kernel(
    float* __restrict__ x, const float* __restrict__ delta,
    const float* __restrict__ g, const float* __restrict__ beta)
{
  const int lane = threadIdx.x & 31;
  const int row  = blockIdx.x * 8 + (threadIdx.x >> 5);
  float* p       = x + (size_t)row * D_;
  const float* d = delta + (size_t)row * D_;
  float v[16];
  float sum = 0.f;
  #pragma unroll
  for (int j = 0; j < 16; ++j) { v[j] = p[lane + 32 * j] + d[lane + 32 * j]; sum += v[j]; }
  #pragma unroll
  for (int o = 16; o >= 1; o >>= 1) sum += __shfl_xor(sum, o, 32);
  float mu = sum * (1.f / (float)D_);
  float var = 0.f;
  #pragma unroll
  for (int j = 0; j < 16; ++j) { float t = v[j] - mu; var += t * t; }
  #pragma unroll
  for (int o = 16; o >= 1; o >>= 1) var += __shfl_xor(var, o, 32);
  float inv = rsqrtf(var * (1.f / (float)D_) + 1e-5f);
  #pragma unroll
  for (int j = 0; j < 16; ++j) {
    int c = lane + 32 * j;
    p[c] = (v[j] - mu) * inv * g[c] + beta[c];
  }
}

// ---------------------------------------------------------------------------
// Host-side orchestration.
// Input flattening assumption (recursive dict-insertion order, 49 leaves):
//  0 src  1 tgt  2 tgt_causal_mask  3 src_emb  4 tgt_emb
//  enc: 5 Wq 6 bq 7 Wk 8 bk 9 Wv 10 bv 11 Wo 12 bo
//       13 W1 14 b1 15 W2 16 b2 17 ln1_g 18 ln1_b 19 ln2_g 20 ln2_b
//  dec: 21..28 self_attn (Wq bq Wk bk Wv bv Wo bo)
//       29..36 cross_attn (Wq bq Wk bk Wv bv Wo bo)
//       37 W1 38 b1 39 W2 40 b2 41 ln1_g 42 ln1_b 43 ln2_g 44 ln2_b 45 ln3_g 46 ln3_b
//  47 Wout  48 bout
// ---------------------------------------------------------------------------
extern "C" void kernel_launch(void* const* d_in, const int* in_sizes, int n_in,
                              void* d_out, int out_size, void* d_ws, size_t ws_size,
                              hipStream_t stream)
{
  const int*   src     = (const int*)d_in[0];
  const int*   tgt     = (const int*)d_in[1];
  const float* src_emb = (const float*)d_in[3];
  const float* tgt_emb = (const float*)d_in[4];
  const float* F[49];
  for (int i = 0; i < n_in && i < 49; ++i) F[i] = (const float*)d_in[i];
  const float* Wout = F[47];
  const float* bout = F[48];

  // workspace carve-up (all f32)
  float* ws = (float*)d_ws;
  const size_t A = (size_t)NTOK * D_;
  float* x      = ws;                 // encoder activations [S,B,D]
  float* y      = x + A;              // decoder activations [T,B,D]
  float* q      = y + A;
  float* k      = q + A;
  float* v      = k + A;
  float* tmp    = v + A;              // attn-out / ffn-out
  float* ctx    = tmp + A;
  float* hidden = ctx + A;            // [NTOK, DFF]
  float* scores = hidden + (size_t)NTOK * DFF_;  // [BH, S, S]

  auto gemm = [&](const float* Ain, const float* W, const float* bias,
                  float* Cout, int M, int N, int K, int relu) {
    gemm_bias_kernel<<<dim3(N / 64, M / 128), 256, 0, stream>>>(Ain, W, bias, Cout, M, N, K, relu);
  };

  auto mha = [&](const float* xq, const float* xkv, const float* const* P,
                 int causal, float* outbuf, size_t wOff, size_t bOff) {
    // P: {Wq,bq,Wk,bk,Wv,bv,Wo,bo}
    gemm(xq,  P[0] + wOff, P[1] + bOff, q, NTOK, D_, D_, 0);
    gemm(xkv, P[2] + wOff, P[3] + bOff, k, NTOK, D_, D_, 0);
    gemm(xkv, P[4] + wOff, P[5] + bOff, v, NTOK, D_, D_, 0);
    attn_scores_kernel<<<dim3(S_ / 16, S_ / 16, BH_), 32, 0, stream>>>(q, k, scores, causal);
    softmax_kernel<<<(BH_ * S_) / 8, 256, 0, stream>>>(scores);
    attn_ctx_kernel<<<dim3(DK_ / 16, S_ / 16, BH_), 32, 0, stream>>>(scores, v, ctx);
    gemm(ctx, P[6] + wOff, P[7] + bOff, outbuf, NTOK, D_, D_, 0);
  };

  // embeddings + positional encoding
  embed_kernel<<<(NTOK * D_) / 256, 256, 0, stream>>>(src, src_emb, x);
  embed_kernel<<<(NTOK * D_) / 256, 256, 0, stream>>>(tgt, tgt_emb, y);

  // ---------------- encoder ----------------
  const float* encAttn[8] = {F[5], F[6], F[7], F[8], F[9], F[10], F[11], F[12]};
  for (int i = 0; i < NE_; ++i) {
    const size_t wDD = (size_t)i * D_ * D_;
    const size_t bD  = (size_t)i * D_;
    const size_t wDF = (size_t)i * D_ * DFF_;
    const size_t bF  = (size_t)i * DFF_;
    mha(x, x, encAttn, /*causal=*/0, tmp, wDD, bD);
    add_ln_kernel<<<NTOK / 8, 256, 0, stream>>>(x, tmp, F[17] + bD, F[18] + bD);
    gemm(x, F[13] + wDF, F[14] + bF, hidden, NTOK, DFF_, D_, /*relu=*/1);
    gemm(hidden, F[15] + wDF, F[16] + bD, tmp, NTOK, D_, DFF_, 0);
    add_ln_kernel<<<NTOK / 8, 256, 0, stream>>>(x, tmp, F[19] + bD, F[20] + bD);
  }

  // ---------------- decoder ----------------
  const float* selfAttn[8]  = {F[21], F[22], F[23], F[24], F[25], F[26], F[27], F[28]};
  const float* crossAttn[8] = {F[29], F[30], F[31], F[32], F[33], F[34], F[35], F[36]};
  for (int i = 0; i < ND_; ++i) {
    const size_t wDD = (size_t)i * D_ * D_;
    const size_t bD  = (size_t)i * D_;
    const size_t wDF = (size_t)i * D_ * DFF_;
    const size_t bF  = (size_t)i * DFF_;
    mha(y, y, selfAttn, /*causal=*/1, tmp, wDD, bD);
    add_ln_kernel<<<NTOK / 8, 256, 0, stream>>>(y, tmp, F[41] + bD, F[42] + bD);
    mha(y, x, crossAttn, /*causal=*/0, tmp, wDD, bD);
    add_ln_kernel<<<NTOK / 8, 256, 0, stream>>>(y, tmp, F[43] + bD, F[44] + bD);
    gemm(y, F[37] + wDF, F[38] + bF, hidden, NTOK, DFF_, D_, /*relu=*/1);
    gemm(hidden, F[39] + wDF, F[40] + bD, tmp, NTOK, D_, DFF_, 0);
    add_ln_kernel<<<NTOK / 8, 256, 0, stream>>>(y, tmp, F[45] + bD, F[46] + bD);
  }

  // ---------------- output projection: [T*B, D] @ [D, V] + bout ----------------
  gemm(y, Wout, bout, (float*)d_out, NTOK, V_, D_, 0);
}
